// Attention_6373731468040
// MI455X (gfx1250) — matrix operand order repaired and checked by reading, not compiled, not run on hardware
//
#include <hip/hip_runtime.h>

typedef __attribute__((ext_vector_type(16))) _Float16 v16h;
typedef __attribute__((ext_vector_type(8)))  _Float16 v8h;
typedef __attribute__((ext_vector_type(4)))  _Float16 v4h;
typedef __attribute__((ext_vector_type(8)))  float    v8f;

#define NB    256   // nodes
#define FIN   128
#define FOUT  128
#define MID   192
#define HSTR  200   // padded LDS row stride in halves

// Branch-free tanh: native v_tanh_f32 on gfx1250 if the builtin exists,
// else clamp + v_exp_f32 + v_rcp_f32 identity (no exec divergence).
__device__ __forceinline__ float fast_tanh(float x) {
#if __has_builtin(__builtin_amdgcn_tanhf)
    return __builtin_amdgcn_tanhf(x);
#elif __has_builtin(__builtin_amdgcn_tanh_f32)
    return __builtin_amdgcn_tanh_f32(x);
#else
    float xc = fminf(fmaxf(x, -10.f), 10.f);
    float e  = __builtin_amdgcn_exp2f(xc * 2.8853900817779268f); // 2*log2(e)
    return (e - 1.f) * __builtin_amdgcn_rcpf(e + 1.f);
#endif
}

// ---------------------------------------------------------------------------
// Phase 1: A12[r][0:192]   = inp[r] @ W1[0:128,:]   + b1   (r = b*256+n)
//          A12[r][192:384] = inp[r] @ W1[128:256,:]
// ~0.1 GFLOP total - plain f32 FMA.
// ---------------------------------------------------------------------------
__global__ void pair_mlp_phase1(const float* __restrict__ inp,
                                const float* __restrict__ W1,
                                const float* __restrict__ b1,
                                float* __restrict__ A12) {
    int idx = blockIdx.x * blockDim.x + threadIdx.x;   // 1024*384 threads
    int r = idx / 384;
    int c = idx - r * 384;
    int which = c / MID;          // 0 -> A1 (+b1), 1 -> A2
    int k = c - which * MID;
    const float* xrow = inp + (size_t)r * FIN;
    const float* wcol = W1 + (size_t)which * FIN * MID + k;
    float s = 0.f;
#pragma unroll 8
    for (int t = 0; t < FIN; ++t) s += xrow[t] * wcol[(size_t)t * MID];
    if (which == 0) s += b1[k];
    A12[(size_t)r * 384 + c] = s;
}

// ---------------------------------------------------------------------------
// Phase 2: one workgroup per (b,i). 8 waves; wave w owns output cols w*16..+15.
// Per 16-j tile: build h = relu(A1[i]+b1+A2[j]) as f16 in LDS (shared),
// 6x v_wmma_f32_16x16x32_f16 per wave (K=192), native tanh, mask j==i,
// accumulate; final row reduction via C-layout + shfl_xor(16).
// ---------------------------------------------------------------------------
__global__ __launch_bounds__(256, 1)
void pair_mlp_phase2(const float* __restrict__ A12,
                     const float* __restrict__ W2,
                     const float* __restrict__ b2,
                     float* __restrict__ out) {
    __shared__ float a1s[MID];
    __shared__ __align__(16) _Float16 hl[16 * HSTR];

    const int tid  = threadIdx.x;
    const int lane = tid & 31;
    const int wv   = tid >> 5;       // 0..7 : n-tile index
    const int bi   = blockIdx.x;     // b*256 + i
    const int i    = bi & (NB - 1);
    const int hi16 = lane >> 4;      // 0 | 1 (lane half)
    const int l15  = lane & 15;

    // --- preload B fragments: W2 cols [wv*16, wv*16+16), f32 -> f16.
    // B k order chosen here (contiguous, same as A; not AMD's table order): lane = col; lanes 0-15 K=s*32+0..15,
    // lanes 16-31 K=s*32+16..31, packed in K order.
    const int ncol = wv * 16 + l15;
    v16h bfrag[6];
#pragma unroll
    for (int s = 0; s < 6; ++s) {
        const int kb = s * 32 + hi16 * 16;
#pragma unroll
        for (int h = 0; h < 16; ++h)
            bfrag[s][h] = (_Float16)W2[(size_t)(kb + h) * FOUT + ncol];
    }
    const float b2n = b2[ncol];

    // --- stage A1[i]+b1 row into LDS
    const float* a1g = A12 + (size_t)bi * 384;
    for (int k = tid; k < MID; k += 256) a1s[k] = a1g[k];

    float acc[8];
#pragma unroll
    for (int g = 0; g < 8; ++g) acc[g] = 0.f;

    const float* A2base = A12 + (size_t)(bi & ~(NB - 1)) * 384 + MID;

    // uniform h-tile mapping: thread -> row (tid>>4), 12 consecutive k
    const int hjj = tid >> 4;            // 0..15
    const int hk0 = (tid & 15) * 12;     // 0..180, 48B-aligned
    _Float16* hdst = hl + hjj * HSTR + hk0;

    __syncthreads();

    for (int jt = 0; jt < 16; ++jt) {
        // build h tile: relu(A1[i]+b1 + A2[j]) -> f16, packed 4-half stores
        {
            const float* a2r = A2base + (size_t)(jt * 16 + hjj) * 384 + hk0;
            const float* a1r = a1s + hk0;
#pragma unroll
            for (int c4 = 0; c4 < 3; ++c4) {
                v4h o;
#pragma unroll
                for (int u = 0; u < 4; ++u) {
                    float v = a1r[c4 * 4 + u] + a2r[c4 * 4 + u];
                    o[u] = (_Float16)(v > 0.f ? v : 0.f);
                }
                *(v4h*)(hdst + c4 * 4) = o;
            }
        }
        __syncthreads();

        // A k order chosen here (contiguous, same as B; not AMD's table order): lane = row; per k-step s:
        // halves 0..7  = K = s*32 + hi16*16 + 0..7
        // halves 8..15 = K = s*32 + hi16*16 + 8..15
        v16h afrag[6];
#pragma unroll
        for (int s = 0; s < 6; ++s) {
            const _Float16* ap = hl + l15 * HSTR + s * 32 + hi16 * 16;
            v8h lo = *(const v8h*)ap;
            v8h hp = *(const v8h*)(ap + 8);
            afrag[s] = __builtin_shufflevector(lo, hp,
                           0,1,2,3,4,5,6,7,8,9,10,11,12,13,14,15);
        }

        v8f c = {};
#pragma unroll
        for (int s = 0; s < 6; ++s)
            c = __builtin_amdgcn_wmma_f32_16x16x32_f16(
                    false, afrag[s], false, bfrag[s], (short)0, c, false, false);

        // C/D layout: VGPR g, lanes 0-15 -> row M=g, lanes 16-31 -> M=g+8
#pragma unroll
        for (int g = 0; g < 8; ++g) {
            int jglob = jt * 16 + g + hi16 * 8;
            float t = fast_tanh(c[g] + b2n);
            acc[g] += (jglob == i) ? 0.f : t;
        }
        __syncthreads();   // before overwriting hl next iteration
    }

    // reduce over the tile's 16 rows: 8 per-lane slots + lane-half merge
    float s = 0.f;
#pragma unroll
    for (int g = 0; g < 8; ++g) s += acc[g];
    s += __shfl_xor(s, 16);
    if (lane < 16) out[(size_t)bi * FOUT + ncol] = s;
}

// ---------------------------------------------------------------------------
extern "C" void kernel_launch(void* const* d_in, const int* in_sizes, int n_in,
                              void* d_out, int out_size, void* d_ws, size_t ws_size,
                              hipStream_t stream) {
    (void)in_sizes; (void)n_in; (void)out_size; (void)ws_size;
    const float* inp = (const float*)d_in[0];   // (4,256,128)
    const float* W1  = (const float*)d_in[1];   // (256,192)
    const float* b1  = (const float*)d_in[2];   // (192,)
    const float* W2  = (const float*)d_in[3];   // (192,128)
    const float* b2  = (const float*)d_in[4];   // (128,)
    float* out = (float*)d_out;                 // (4,256,128)
    float* A12 = (float*)d_ws;                  // 1024*384 f32 = 1.5 MB

    pair_mlp_phase1<<<(1024 * 384) / 256, 256, 0, stream>>>(inp, W1, b1, A12);
    pair_mlp_phase2<<<4 * NB, 256, 0, stream>>>(A12, W2, b2, out);
}
